// MambaTrans_39247411151232
// MI455X (gfx1250) — matrix operand order repaired and checked by reading, not compiled, not run on hardware
//
#include <hip/hip_runtime.h>
#include <hip/hip_bf16.h>

typedef _Float16 v16h __attribute__((ext_vector_type(16)));
typedef _Float16 v8h  __attribute__((ext_vector_type(8)));
typedef _Float16 v4h  __attribute__((ext_vector_type(4)));
typedef float    v8f  __attribute__((ext_vector_type(8)));

#define TOKENS (8 * 8192)
#define LSEQ   8192
#define DBCP   288   // 264 padded to 18 WMMA tiles (NT=3 divides evenly -> no tail)

// ---------------- tile loaders (wave32 WMMA 16x16x32 f16 layouts) ----------------
// A-matrix 16x32 f16: lane L (r = L&15, kh = L>>4) holds row r,
//   halves 0..7  -> K = kh*8 + j        (VGPR0-3)
//   halves 8..15 -> K = 16 + kh*8 + j   (VGPR4-7)
__device__ inline v16h load_a_tile(const _Float16* __restrict__ A, int K,
                                   int m0, int k0, int lane) {
  int r  = lane & 15;
  int kh = lane >> 4;
  const _Float16* p = A + (size_t)(m0 + r) * K + k0 + kh * 8;
  v8h lo = *(const v8h*)p;
  v8h hi = *(const v8h*)(p + 16);
  return __builtin_shufflevector(lo, hi, 0,1,2,3,4,5,6,7,8,9,10,11,12,13,14,15);
}

// B-matrix 32x16 f16: lane L (n = L&15) holds column n,
//   halves 0..7 -> K = (L>>4)*8 + j, halves 8..15 -> K = 16 + (L>>4)*8 + j.  W is row-major (N x K), B[k][n] = W[n][k].
__device__ inline v16h load_b_tile(const _Float16* __restrict__ W, int K,
                                   int n0, int k0, int lane) {
  int n  = lane & 15;
  int kb = (lane >> 4) * 8;
  const _Float16* p = W + (size_t)(n0 + n) * K + k0 + kb;
  v8h lo = *(const v8h*)p;
  v8h hi = *(const v8h*)(p + 16);
  return __builtin_shufflevector(lo, hi, 0,1,2,3,4,5,6,7,8,9,10,11,12,13,14,15);
}

// ---------------- generic WMMA GEMM: out = A(MxK,f16) * W(NxK,f16)^T ----------------
// All shape parameters are compile-time: K-loop and tile-loop fully unroll, grids are
// sized exactly, and there is NO control flow around the WMMAs (EXEC stays all-ones,
// as the ISA requires). One wave computes a 16x(16*NT) strip, reusing each A tile NT x.
// MODE 0: in_proj  -> split cols [0,256) to oh0 (xi), [256,512) to oh1 (z), f16
// MODE 1: x_proj   -> oh0 f16 (dbc, padded N=288)
// MODE 2: out_proj -> of f32 = acc + resid (residual add of original x)
// MODE 3: head     -> of f32 = acc + bias[col] + resid (x1)
template <int MODE, int K, int NTILES, int NT, int NLD>
__global__ __launch_bounds__(256) void gemm16(
    const _Float16* __restrict__ A, const _Float16* __restrict__ W,
    _Float16* __restrict__ oh0, _Float16* __restrict__ oh1,
    float* __restrict__ of, const float* __restrict__ resid,
    const float* __restrict__ bias) {
  constexpr int NGROUPS = NTILES / NT;
  int lane = threadIdx.x & 31;
  int gw   = blockIdx.x * 8 + (threadIdx.x >> 5);
  int mt   = gw / NGROUPS;          // compile-time divisor
  int ng   = gw % NGROUPS;
  int m0   = mt * 16;

  v8f acc[NT];
#pragma unroll
  for (int t = 0; t < NT; ++t) {
    v8f z8 = {0.f, 0.f, 0.f, 0.f, 0.f, 0.f, 0.f, 0.f};
    acc[t] = z8;
  }

#pragma unroll
  for (int k0 = 0; k0 < K; k0 += 32) {
    v16h a = load_a_tile(A, K, m0, k0, lane);
#pragma unroll
    for (int t = 0; t < NT; ++t) {
      v16h b = load_b_tile(W, K, (ng * NT + t) * 16, k0, lane);
      acc[t] = __builtin_amdgcn_wmma_f32_16x16x32_f16(
          false, a, false, b, (short)0, acc[t], false, false);
    }
  }

  // C/D layout: VGPR v, lane L -> row = m0 + v + (L>>4)*8, col = n0 + (L&15)
  int rbase = m0 + (lane >> 4) * 8;
  int cl    = lane & 15;
#pragma unroll
  for (int t = 0; t < NT; ++t) {
    int col = (ng * NT + t) * 16 + cl;
#pragma unroll
    for (int v = 0; v < 8; ++v) {
      int row   = rbase + v;
      float val = acc[t][v];
      if (MODE == 0) {
        if (col < 256) oh0[(size_t)row * 256 + col] = (_Float16)val;
        else           oh1[(size_t)row * 256 + (col - 256)] = (_Float16)val;
      } else if (MODE == 1) {
        oh0[(size_t)row * NLD + col] = (_Float16)val;
      } else if (MODE == 2) {
        size_t o = (size_t)row * NLD + col;
        of[o] = val + resid[o];
      } else {
        size_t o = (size_t)row * NLD + col;
        of[o] = val + bias[col] + resid[o];
      }
    }
  }
}

// ---------------- layernorm over 128 features: one wave per token ----------------
__global__ __launch_bounds__(256) void ln128(const float* __restrict__ x,
                                             const float* __restrict__ g,
                                             const float* __restrict__ b,
                                             _Float16* __restrict__ out) {
  int lane   = threadIdx.x & 31;
  size_t tok = (size_t)blockIdx.x * 8 + (threadIdx.x >> 5);
  const float4 v = *(const float4*)(x + tok * 128 + lane * 4);
  float s = v.x + v.y + v.z + v.w;
  float q = v.x * v.x + v.y * v.y + v.z * v.z + v.w * v.w;
  for (int off = 16; off; off >>= 1) {
    s += __shfl_xor(s, off, 32);
    q += __shfl_xor(q, off, 32);
  }
  float mean = s * (1.f / 128.f);
  float var  = q * (1.f / 128.f) - mean * mean;
  float rs   = rsqrtf(var + 1e-5f);
  float vv[4] = {v.x, v.y, v.z, v.w};
  int i0 = lane * 4;
#pragma unroll
  for (int j = 0; j < 4; ++j)
    out[tok * 128 + i0 + j] = (_Float16)((vv[j] - mean) * rs * g[i0 + j] + b[i0 + j]);
}

// ---------------- causal depthwise conv (4 taps) + silu ----------------
__global__ __launch_bounds__(256) void conv_silu(const _Float16* __restrict__ xi,
                                                 const float* __restrict__ cw,
                                                 const float* __restrict__ cb,
                                                 _Float16* __restrict__ xc) {
  size_t i   = (size_t)blockIdx.x * blockDim.x + threadIdx.x;
  int c      = (int)(i & 255);
  size_t tok = i >> 8;
  int l      = (int)(tok & (LSEQ - 1));
  float acc  = cb[c];
#pragma unroll
  for (int t = 0; t < 4; ++t) {
    int ls = l - 3 + t;
    if (ls >= 0) acc += cw[c * 4 + t] * (float)xi[(tok - 3 + t) * 256 + c];
  }
  float sig = 1.f / (1.f + __expf(-acc));
  xc[i] = (_Float16)(acc * sig);
}

// ---------------- dt = softplus(dbc[:, :8] @ dt_w^T + dt_b) ----------------
__global__ __launch_bounds__(256) void dt_softplus(const _Float16* __restrict__ dbc,
                                                   const float* __restrict__ w,
                                                   const float* __restrict__ b,
                                                   float* __restrict__ dt) {
  size_t i   = (size_t)blockIdx.x * blockDim.x + threadIdx.x;
  int d      = (int)(i & 255);
  size_t tok = i >> 8;
  const _Float16* p = dbc + tok * DBCP;
  float s = b[d];
#pragma unroll
  for (int r = 0; r < 8; ++r) s += (float)p[r] * w[d * 8 + r];
  dt[i] = (s > 20.f) ? s : log1pf(__expf(s));
}

// ---------------- selective scan: one wave per (batch, d_inner) channel ----------
// lane owns 4 states; h kept in registers across all 8192 steps.
// Epilogue fused: y = (scan + D*xc) * silu(z), stored f16 for the out_proj GEMM.
__global__ __launch_bounds__(256) void ssm_scan(const float* __restrict__ dt,
                                                const _Float16* __restrict__ dbc,
                                                const _Float16* __restrict__ xc,
                                                const _Float16* __restrict__ z,
                                                const float* __restrict__ A_log,
                                                const float* __restrict__ Dp,
                                                _Float16* __restrict__ y) {
  int lane = threadIdx.x & 31;
  int wv   = threadIdx.x >> 5;
  int b    = blockIdx.x >> 5;                 // 8 batches
  int d    = ((blockIdx.x & 31) << 3) + wv;   // 256 channels
  int n4   = lane << 2;
  float a0 = -__expf(A_log[d * 128 + n4 + 0]);
  float a1 = -__expf(A_log[d * 128 + n4 + 1]);
  float a2 = -__expf(A_log[d * 128 + n4 + 2]);
  float a3 = -__expf(A_log[d * 128 + n4 + 3]);
  float Dv = Dp[d];
  float h0 = 0.f, h1 = 0.f, h2 = 0.f, h3 = 0.f;
  size_t tok = (size_t)b * LSEQ;
  for (int l = 0; l < LSEQ; ++l, ++tok) {
    float dtv = dt[tok * 256 + d];
    float xv  = (float)xc[tok * 256 + d];
    const _Float16* row = dbc + tok * DBCP;
    v4h Bv = *(const v4h*)(row + 8 + n4);     // dbc cols [8,136)   = B
    v4h Cv = *(const v4h*)(row + 136 + n4);   // dbc cols [136,264) = C
    float dx = dtv * xv;
    h0 = __expf(dtv * a0) * h0 + dx * (float)Bv[0];
    h1 = __expf(dtv * a1) * h1 + dx * (float)Bv[1];
    h2 = __expf(dtv * a2) * h2 + dx * (float)Bv[2];
    h3 = __expf(dtv * a3) * h3 + dx * (float)Bv[3];
    float ys = h0 * (float)Cv[0] + h1 * (float)Cv[1] +
               h2 * (float)Cv[2] + h3 * (float)Cv[3];
    for (int off = 16; off; off >>= 1) ys += __shfl_xor(ys, off, 32);
    if (lane == 0) {
      float zv   = (float)z[tok * 256 + d];
      float gate = zv / (1.f + __expf(-zv));
      y[tok * 256 + d] = (_Float16)((ys + Dv * xv) * gate);
    }
  }
}

// ---------------- weight conversion helpers ----------------
__global__ __launch_bounds__(256) void cvt_f32_f16(const float* __restrict__ s,
                                                   _Float16* __restrict__ d, int n) {
  int i = blockIdx.x * blockDim.x + threadIdx.x;
  if (i < n) d[i] = (_Float16)s[i];
}

// x_proj_w (264x256) -> zero-padded (288x256) f16
__global__ __launch_bounds__(256) void xproj_pad(const float* __restrict__ s,
                                                 _Float16* __restrict__ d) {
  int i = blockIdx.x * blockDim.x + threadIdx.x;
  if (i >= DBCP * 256) return;
  int row = i >> 8;
  d[i] = (row < 264) ? (_Float16)s[i] : (_Float16)0.f;
}

extern "C" void kernel_launch(void* const* d_in, const int* in_sizes, int n_in,
                              void* d_out, int out_size, void* d_ws, size_t ws_size,
                              hipStream_t stream) {
  (void)in_sizes; (void)n_in; (void)out_size; (void)ws_size;
  const float* x       = (const float*)d_in[0];
  const float* n1g     = (const float*)d_in[1];
  const float* n1b     = (const float*)d_in[2];
  const float* n2g     = (const float*)d_in[3];
  const float* n2b     = (const float*)d_in[4];
  const float* in_w    = (const float*)d_in[5];
  const float* conv_w  = (const float*)d_in[6];
  const float* conv_b  = (const float*)d_in[7];
  const float* xproj_w = (const float*)d_in[8];
  const float* dt_w    = (const float*)d_in[9];
  const float* dt_b    = (const float*)d_in[10];
  const float* A_log   = (const float*)d_in[11];
  const float* Dp      = (const float*)d_in[12];
  const float* out_w   = (const float*)d_in[13];
  const float* head_w  = (const float*)d_in[14];
  const float* head_b  = (const float*)d_in[15];
  float* out = (float*)d_out;

  char* ws = (char*)d_ws;
  size_t off = 0;
  auto alloc = [&](size_t bytes) -> char* {
    char* p = ws + off;
    off += (bytes + 255) & ~(size_t)255;
    return p;
  };
  _Float16* u16   = (_Float16*)alloc((size_t)TOKENS * 128 * 2); // ln1 out; reused as ln2 out
  _Float16* xi16  = (_Float16*)alloc((size_t)TOKENS * 256 * 2); // xi; reused as y after conv
  _Float16* z16   = (_Float16*)alloc((size_t)TOKENS * 256 * 2);
  _Float16* xc16  = (_Float16*)alloc((size_t)TOKENS * 256 * 2);
  _Float16* dbc16 = (_Float16*)alloc((size_t)TOKENS * DBCP * 2);
  float*    dt32  = (float*)   alloc((size_t)TOKENS * 256 * 4);
  float*    x1    = (float*)   alloc((size_t)TOKENS * 128 * 4);
  _Float16* wi16  = (_Float16*)alloc((size_t)512 * 128 * 2);
  _Float16* wx16  = (_Float16*)alloc((size_t)DBCP * 256 * 2);
  _Float16* wo16  = (_Float16*)alloc((size_t)128 * 256 * 2);
  _Float16* wh16  = (_Float16*)alloc((size_t)128 * 128 * 2);

  // weights -> f16 (tiny; stays resident in L2)
  cvt_f32_f16<<<(512 * 128 + 255) / 256, 256, 0, stream>>>(in_w, wi16, 512 * 128);
  xproj_pad<<<(DBCP * 256 + 255) / 256, 256, 0, stream>>>(xproj_w, wx16);
  cvt_f32_f16<<<(128 * 256 + 255) / 256, 256, 0, stream>>>(out_w, wo16, 128 * 256);
  cvt_f32_f16<<<(128 * 128 + 255) / 256, 256, 0, stream>>>(head_w, wh16, 128 * 128);

  const int Mt = TOKENS / 16;  // 4096 row-tiles

  // 1) layernorm1
  ln128<<<TOKENS / 8, 256, 0, stream>>>(x, n1g, n1b, u16);
  // 2) in_proj (K=128, 32 tiles, NT=4 -> 8 groups) -> xi, z     [exact grid: 4096 blocks]
  gemm16<0, 128, 32, 4, 512><<<Mt * 8 / 8, 256, 0, stream>>>(
      u16, wi16, xi16, z16, nullptr, nullptr, nullptr);
  // 3) depthwise conv + silu -> xc
  conv_silu<<<(TOKENS * 256) / 256, 256, 0, stream>>>(xi16, conv_w, conv_b, xc16);
  // 4) x_proj (K=256, 18 tiles, NT=3 -> 6 groups) -> dbc        [exact grid: 3072 blocks]
  gemm16<1, 256, 18, 3, DBCP><<<Mt * 6 / 8, 256, 0, stream>>>(
      xc16, wx16, dbc16, nullptr, nullptr, nullptr, nullptr);
  // 5) dt projection + softplus
  dt_softplus<<<(TOKENS * 256) / 256, 256, 0, stream>>>(dbc16, dt_w, dt_b, dt32);
  // 6) selective scan (+ D*xc, * silu(z)) -> y (reuses xi16)
  ssm_scan<<<256, 256, 0, stream>>>(dt32, dbc16, xc16, z16, A_log, Dp, xi16);
  // 7) out_proj (K=256, 8 tiles, NT=4 -> 2 groups) + resid x -> x1  [1024 blocks]
  gemm16<2, 256, 8, 4, 128><<<Mt * 2 / 8, 256, 0, stream>>>(
      xi16, wo16, nullptr, nullptr, x1, x, nullptr);
  // 8) layernorm2 (reuse u16)
  ln128<<<TOKENS / 8, 256, 0, stream>>>(x1, n2g, n2b, u16);
  // 9) head (K=128, 8 tiles, NT=4 -> 2 groups) + bias + resid x1 -> out [1024 blocks]
  gemm16<3, 128, 8, 4, 128><<<Mt * 2 / 8, 256, 0, stream>>>(
      u16, wh16, nullptr, nullptr, out, x1, head_b);
}
